// Scaled_Dot_Product_Attention_2783138808318
// MI455X (gfx1250) — matrix operand order repaired and checked by reading, not compiled, not run on hardware
//
#include <hip/hip_runtime.h>
#include <hip/hip_bf16.h>

// ---- problem dims ----
#define BATCH  32
#define SEQ    1024
#define HIDDEN 1024
#define HEAD   128

typedef __attribute__((ext_vector_type(16))) __bf16 v16bf;
typedef __attribute__((ext_vector_type(8)))  __bf16 v8bf;
typedef __attribute__((ext_vector_type(8)))  float  v8f;
typedef __attribute__((ext_vector_type(4)))  float  v4f;

__device__ __forceinline__ v8f vzero8() {
  v8f z = {0.f, 0.f, 0.f, 0.f, 0.f, 0.f, 0.f, 0.f};
  return z;
}

// D = A(16x32 bf16) * B(32x16 bf16) + C(16x16 f32)
__device__ __forceinline__ v8f wmma_bf16(v16bf a, v16bf b, v8f c) {
  return __builtin_amdgcn_wmma_f32_16x16x32_bf16(
      /*neg_a=*/false, a, /*neg_b=*/false, b,
      /*c_mod=*/(short)0, c, /*reuse_a=*/false, /*reuse_b=*/false);
}

// ---------------------------------------------------------------------------
// Kernel W: convert Wq/Wk/Wv (fp32 [1024,128]) into bf16 B-fragment layout:
//   wsw[mat][kchunk(32)][ntile(8)][lane(32)][t(16)]
// B-frag element t at lane l = W[kchunk*32 + (t<8 ? 8*(l>>4)+t : 16+8*(l>>4)+(t-8))][ntile*16 + (l&15)]
// One thread produces one 16-element lane fragment (32B contiguous store).
// ---------------------------------------------------------------------------
__global__ void __launch_bounds__(256)
swizzle_w_kernel(const float* __restrict__ Wq, const float* __restrict__ Wk,
                 const float* __restrict__ Wv, __bf16* __restrict__ wsw) {
  int tid = blockIdx.x * blockDim.x + threadIdx.x;     // 3*32*8*32 = 24576
  int lane   = tid & 31;
  int ntile  = (tid >> 5) & 7;
  int kchunk = (tid >> 8) & 31;
  int mat    = tid >> 13;
  const float* W = (mat == 0) ? Wq : (mat == 1) ? Wk : Wv;
  int h = lane >> 4, n = lane & 15;
  v16bf frag;
#pragma unroll
  for (int t = 0; t < 16; ++t)
    frag[t] = (__bf16)W[(kchunk * 32 + (t < 8 ? 8 * h + t : 16 + 8 * h + (t - 8))) * HEAD + ntile * 16 + n];
  *(v16bf*)(wsw + ((size_t)(((mat * 32 + kchunk) * 8 + ntile) * 32 + lane) << 4)) = frag;
}

// ---------------------------------------------------------------------------
// Kernel P: QKV projection. grid = (512, 3), block = 128 (4 waves).
// Wave handles strip gs = blockIdx.x*4 + wave: rows [gs*16, gs*16+16) of the
// flattened [32768, 1024] X, times W[mat] -> 16x128, K=1024 -> 256 WMMAs.
// mat 0 -> q row-major bf16; mat 1 -> k row-major bf16; mat 2 -> v swizzled
// into out-GEMM B-fragment layout vsw[b][kchunk(32)][ntile(8)][lane(32)][t(16)].
// ---------------------------------------------------------------------------
__global__ void __launch_bounds__(128)
qkv_proj_kernel(const float* __restrict__ X,
                const float* __restrict__ bq, const float* __restrict__ bk,
                const float* __restrict__ bvv,
                const __bf16* __restrict__ wsw,
                __bf16* __restrict__ qb, __bf16* __restrict__ kb,
                __bf16* __restrict__ vsw) {
  const int lane = threadIdx.x & 31;
  const int wid  = threadIdx.x >> 5;
  const int mat  = blockIdx.y;
  const int gs   = blockIdx.x * 4 + wid;   // [0, 2048)
  const int row0 = gs * 16;
  const int h = lane >> 4, n = lane & 15;

  const __bf16* wmat = wsw + (size_t)mat * (32 * 8 * 32 * 16);
  const float*  xrow = X + (size_t)(row0 + n) * HIDDEN;

  v8f acc[8];
#pragma unroll
  for (int j = 0; j < 8; ++j) acc[j] = vzero8();

  for (int kc = 0; kc < HIDDEN; kc += 32) {
    // A fragment (16-bit A layout): t<8 -> k = kc+8h+t ; t>=8 -> k = kc+16+8h+(t-8)
    v4f a0 = *(const v4f*)(xrow + kc + 8 * h);
    v4f a1 = *(const v4f*)(xrow + kc + 8 * h + 4);
    v4f a2 = *(const v4f*)(xrow + kc + 16 + 8 * h);
    v4f a3 = *(const v4f*)(xrow + kc + 16 + 8 * h + 4);
    v16bf af;
#pragma unroll
    for (int i = 0; i < 4; ++i) {
      af[i]      = (__bf16)a0[i];
      af[4 + i]  = (__bf16)a1[i];
      af[8 + i]  = (__bf16)a2[i];
      af[12 + i] = (__bf16)a3[i];
    }
    const __bf16* wchunk = wmat + (size_t)(kc >> 5) * (8 * 32 * 16);
#pragma unroll
    for (int j = 0; j < 8; ++j) {
      v16bf bf = *(const v16bf*)(wchunk + ((j * 32 + lane) << 4));
      acc[j] = wmma_bf16(af, bf, acc[j]);
    }
  }

  const float* bias = (mat == 0) ? bq : (mat == 1) ? bk : bvv;
  if (mat < 2) {
    __bf16* dst = (mat == 0) ? qb : kb;
#pragma unroll
    for (int j = 0; j < 8; ++j) {
      float bval = bias[j * 16 + n];
#pragma unroll
      for (int r = 0; r < 8; ++r) {
        int row = row0 + r + 8 * h;                 // D layout: m = r + 8h, col lane n
        dst[(size_t)row * HEAD + j * 16 + n] = (__bf16)(acc[j][r] + bval);
      }
    }
  } else {
    int b   = row0 >> 10;
    int gsl = gs & 63;          // strip within batch
    int c   = gsl >> 1;         // 32-row chunk
    int hs  = gsl & 1;          // which 16-row half of the chunk
#pragma unroll
    for (int j = 0; j < 8; ++j) {
      float bval = bias[j * 16 + n];
#pragma unroll
      for (int r = 0; r < 8; ++r) {
        int t = r + 8 * h;      // row within 16-row half == consumer frag element
        vsw[((size_t)(((b * 32 + c) * 8 + j) * 32 + (hs * 16 + n)) << 4) + t] =
            (__bf16)(acc[j][r] + bval);
      }
    }
  }
}

// ---------------------------------------------------------------------------
// Kernel S: scores + softmax + probs@V. grid = 32*32 blocks, block = 64 (2 waves).
// Wave handles 16 query rows vs all 1024 keys.
//   pass1: scores via WMMA -> row max (no exp)
//   pass2: recompute scores, p = exp(s - max), accumulate rowsum,
//          stash unnormalized p as bf16 in per-wave LDS (16x1024)
//   pass3: write fp32 normalized probs; out = (p @ V) * (1/rowsum)
// ---------------------------------------------------------------------------
__global__ void __launch_bounds__(64)
attn_kernel(const __bf16* __restrict__ qb, const __bf16* __restrict__ kb,
            const __bf16* __restrict__ vsw,
            float* __restrict__ outp, float* __restrict__ probs) {
  __shared__ __bf16 plds[2][16][SEQ];                 // 64 KB

  const int lane = threadIdx.x & 31;
  const int wid  = threadIdx.x >> 5;
  const int b    = blockIdx.x >> 5;
  const int qblk = blockIdx.x & 31;
  const int m0   = qblk * 32 + wid * 16;              // local q-row base
  const int h = lane >> 4, n = lane & 15;
  const float scale = 0.08838834764831845f;           // 1/sqrt(128)

  // q A-fragments over the 4 d-chunks (rows m0+n, contiguous 16B runs)
  const __bf16* qrow = qb + (size_t)(b * SEQ + m0 + n) * HEAD;
  v16bf qf[4];
#pragma unroll
  for (int dc = 0; dc < 4; ++dc) {
    v8bf lo = *(const v8bf*)(qrow + dc * 32 + 16 * h);
    v8bf hi = *(const v8bf*)(qrow + dc * 32 + 16 * h + 8);
#pragma unroll
    for (int i = 0; i < 8; ++i) { qf[dc][i] = lo[i]; qf[dc][8 + i] = hi[i]; }
  }

  const __bf16* kbase = kb + (size_t)b * SEQ * HEAD;

  // ---- pass 1: row max ----
  float rmax[8];
#pragma unroll
  for (int r = 0; r < 8; ++r) rmax[r] = -3.0e38f;

  for (int kt = 0; kt < 64; ++kt) {
    // k B-frag: element t = k[kt*16 + n][dc*32 + 16h + t]  (32B contiguous)
    const __bf16* krow = kbase + (size_t)(kt * 16 + n) * HEAD + 16 * h;
    v8f acc = vzero8();
#pragma unroll
    for (int dc = 0; dc < 4; ++dc) {
      v16bf kf = *(const v16bf*)(krow + dc * 32);
      acc = wmma_bf16(qf[dc], kf, acc);
    }
#pragma unroll
    for (int r = 0; r < 8; ++r) rmax[r] = fmaxf(rmax[r], acc[r] * scale);
  }
  // reduce across the 16 column-lanes of each half (rows r+8h live per half)
#pragma unroll
  for (int r = 0; r < 8; ++r) {
#pragma unroll
    for (int m = 1; m < 16; m <<= 1)
      rmax[r] = fmaxf(rmax[r], __shfl_xor(rmax[r], m, 32));
  }

  // ---- pass 2: exp, row sum, stash unnormalized probs in LDS ----
  float rsum[8];
#pragma unroll
  for (int r = 0; r < 8; ++r) rsum[r] = 0.f;

  for (int kt = 0; kt < 64; ++kt) {
    const __bf16* krow = kbase + (size_t)(kt * 16 + n) * HEAD + 16 * h;
    v8f acc = vzero8();
#pragma unroll
    for (int dc = 0; dc < 4; ++dc) {
      v16bf kf = *(const v16bf*)(krow + dc * 32);
      acc = wmma_bf16(qf[dc], kf, acc);
    }
#pragma unroll
    for (int r = 0; r < 8; ++r) {
      float p = __expf(acc[r] * scale - rmax[r]);
      rsum[r] += p;
      plds[wid][r + 8 * h][kt * 16 + n] = (__bf16)p;
    }
  }
#pragma unroll
  for (int r = 0; r < 8; ++r) {
#pragma unroll
    for (int m = 1; m < 16; m <<= 1) rsum[r] += __shfl_xor(rsum[r], m, 32);
  }
  float rinv[8];
#pragma unroll
  for (int r = 0; r < 8; ++r) rinv[r] = 1.0f / rsum[r];

  // ---- pass 3a: normalized fp32 probs to global ----
  float* pout = probs + (size_t)(b * SEQ + m0) * SEQ;
#pragma unroll
  for (int r = 0; r < 8; ++r) {
    float* prow = pout + (size_t)(r + 8 * h) * SEQ + n;
    float iv = rinv[r];
    for (int kt = 0; kt < 64; ++kt)
      prow[kt * 16] = (float)plds[wid][r + 8 * h][kt * 16 + n] * iv;
  }

  // ---- pass 3b: out = (p @ V) * rinv ----
  v8f oacc[8];
#pragma unroll
  for (int j = 0; j < 8; ++j) oacc[j] = vzero8();

  const __bf16* vb = vsw + (size_t)b * (32 * 8 * 32 * 16);
  for (int kc = 0; kc < 32; ++kc) {
    // probs A-frag from LDS: two contiguous 16B runs per lane
    v8bf lo = *(const v8bf*)(&plds[wid][n][kc * 32 + 16 * h]);
    v8bf hi = *(const v8bf*)(&plds[wid][n][kc * 32 + 16 * h + 8]);
    v16bf af;
#pragma unroll
    for (int i = 0; i < 8; ++i) { af[i] = lo[i]; af[8 + i] = hi[i]; }
    const __bf16* vchunk = vb + (size_t)kc * (8 * 32 * 16);
#pragma unroll
    for (int j = 0; j < 8; ++j) {
      v16bf vf = *(const v16bf*)(vchunk + ((j * 32 + lane) << 4));
      oacc[j] = wmma_bf16(af, vf, oacc[j]);
    }
  }
  float* orow = outp + (size_t)(b * SEQ + m0) * HEAD;
#pragma unroll
  for (int j = 0; j < 8; ++j) {
#pragma unroll
    for (int r = 0; r < 8; ++r)
      orow[(size_t)(r + 8 * h) * HEAD + j * 16 + n] = oacc[j][r] * rinv[r];
  }
}

// ---------------------------------------------------------------------------
// Workspace layout (bytes):
//   [0,     768K)  swizzled bf16 weights (3 x 32 x 8 x 32 x 16)
//   [1M,    9M)    q bf16 row-major [32*1024, 128]
//   [9M,   17M)    k bf16 row-major [32*1024, 128]
//   [17M,  25M)    v bf16 swizzled  [32][32][8][32][16]
// ---------------------------------------------------------------------------
extern "C" void kernel_launch(void* const* d_in, const int* in_sizes, int n_in,
                              void* d_out, int out_size, void* d_ws, size_t ws_size,
                              hipStream_t stream) {
  (void)in_sizes; (void)n_in; (void)out_size; (void)ws_size;
  const float* X  = (const float*)d_in[0];
  const float* Wq = (const float*)d_in[1];
  const float* bq = (const float*)d_in[2];
  const float* Wk = (const float*)d_in[3];
  const float* bk = (const float*)d_in[4];
  const float* Wv = (const float*)d_in[5];
  const float* bv = (const float*)d_in[6];

  float* out   = (float*)d_out;                              // [32,1024,128]
  float* probs = out + (size_t)BATCH * SEQ * HEAD;           // [32,1024,1024]

  char* ws = (char*)d_ws;
  __bf16* wsw = (__bf16*)(ws);
  __bf16* qb  = (__bf16*)(ws + (1u << 20));
  __bf16* kb  = (__bf16*)(ws + (9u << 20));
  __bf16* vsw = (__bf16*)(ws + (17u << 20));

  swizzle_w_kernel<<<96, 256, 0, stream>>>(Wq, Wk, Wv, wsw);
  qkv_proj_kernel<<<dim3(512, 3), 128, 0, stream>>>(X, bq, bk, bv, wsw, qb, kb, vsw);
  attn_kernel<<<BATCH * (SEQ / 32), 64, 0, stream>>>(qb, kb, vsw, out, probs);
}
